// SparseExpertFFN_22265110463252
// MI455X (gfx1250) — matrix operand order repaired and checked by reading, not compiled, not run on hardware
//
#include <hip/hip_runtime.h>
#include <math.h>

// ---------------------------------------------------------------------------
// Soft-MoE FFN for MI455X (gfx1250).
// Memory-bound (~1.17 TFLOP vs 23.3 TB/s). bf16 staging, TDM->LDS double
// buffered B tiles, v_wmma_f32_16x16x32_bf16 mainloops. Router weight folded
// into h so all experts + shared branch collapse into ONE K=9216 GEMM that
// writes the fp32 output exactly once (no read-modify-write accumulation).
// B fragments explicitly double-buffered so ds_load latency hides behind WMMA.
// ---------------------------------------------------------------------------

typedef __bf16 bf16_t;
typedef __attribute__((ext_vector_type(8)))  float        v8f;
typedef __attribute__((ext_vector_type(4)))  float        v4f;
typedef __attribute__((ext_vector_type(16))) bf16_t       v16bf;
typedef __attribute__((ext_vector_type(8)))  bf16_t       v8bf;
typedef __attribute__((ext_vector_type(4)))  unsigned int u32x4;
typedef __attribute__((ext_vector_type(8)))  int          i32x8;
typedef __attribute__((ext_vector_type(4)))  int          i32x4;

#define NTOK 8192          // B*S
#define DIM  2048          // D
#define NEXP 8             // E
#define HID  1024          // per-expert hidden == shared hidden (DS)
#define KALL (NEXP * HID + HID)   // 9216: stacked K (8 experts + shared)
#define LBW  0.01f

// LDS B-tile: 128 N-rows x 64 K bf16 = 128B/row, TDM pads +16B/row
// (pad_interval=32 dwords, pad_amount=4 dwords) -> 144B stride, bank-safe.
#define KC    64
#define BROW  144
#define BUFB  (128 * BROW)       // 18432 B per buffer
#define SMEMB (2 * BUFB)         // 36864 B double-buffered

// ---------------------------------------------------------------------------
// fp32 -> bf16 elementwise convert (8 elems / thread)
// ---------------------------------------------------------------------------
__global__ void moe_cvt_bf16(const float* __restrict__ in,
                             bf16_t* __restrict__ out, int n) {
  int i = (blockIdx.x * blockDim.x + threadIdx.x) * 8;
  if (i >= n) return;
  v4f a = *(const v4f*)(in + i);
  v4f b = *(const v4f*)(in + i + 4);
  v8bf o;
  o[0] = (bf16_t)a[0]; o[1] = (bf16_t)a[1]; o[2] = (bf16_t)a[2]; o[3] = (bf16_t)a[3];
  o[4] = (bf16_t)b[0]; o[5] = (bf16_t)b[1]; o[6] = (bf16_t)b[2]; o[7] = (bf16_t)b[3];
  *(v8bf*)(out + i) = o;
}

// ---------------------------------------------------------------------------
// Tiled transpose + convert: in (batch, R, C) fp32 -> bf16 at
// out[z*obatch + (c)*ostride + r]  (generalized for column-interleaved W2).
// ---------------------------------------------------------------------------
__global__ void moe_xpose_cvt(const float* __restrict__ in,
                              bf16_t* __restrict__ out, int R, int C,
                              long ostride, long obatch) {
  __shared__ float tile[32][33];
  long ibase = (long)blockIdx.z * (long)R * (long)C;
  long obase = (long)blockIdx.z * obatch;
  int c0 = blockIdx.x * 32, r0 = blockIdx.y * 32;
  int tx = threadIdx.x, ty = threadIdx.y;    // 32 x 8
#pragma unroll
  for (int i = 0; i < 4; ++i) {
    int rr = ty + i * 8;
    tile[rr][tx] = in[ibase + (long)(r0 + rr) * C + c0 + tx];
  }
  __syncthreads();
#pragma unroll
  for (int i = 0; i < 4; ++i) {
    int rr = ty + i * 8;
    out[obase + (long)(c0 + rr) * ostride + r0 + tx] = (bf16_t)tile[tx][rr];
  }
}

// ---------------------------------------------------------------------------
// Router: softmax(x @ Wr + br), one wave32 per token; atomic usage sums.
// ---------------------------------------------------------------------------
__global__ void moe_router(const float* __restrict__ x,
                           const float* __restrict__ Wr,
                           const float* __restrict__ br,
                           float* __restrict__ rw,
                           float* __restrict__ usage) {
  int wave = threadIdx.x >> 5;
  int lane = threadIdx.x & 31;
  int t = blockIdx.x * 8 + wave;
  const float* xr = x + (long)t * DIM;
  float acc[NEXP];
#pragma unroll
  for (int e = 0; e < NEXP; ++e) acc[e] = 0.f;
  for (int d = lane; d < DIM; d += 32) {
    float xv = xr[d];
    const float* w = Wr + (long)d * NEXP;
#pragma unroll
    for (int e = 0; e < NEXP; ++e) acc[e] += xv * w[e];
  }
#pragma unroll
  for (int m = 16; m >= 1; m >>= 1) {
#pragma unroll
    for (int e = 0; e < NEXP; ++e) acc[e] += __shfl_xor(acc[e], m, 32);
  }
  if (lane == 0) {
    float mx = -1e30f;
#pragma unroll
    for (int e = 0; e < NEXP; ++e) { acc[e] += br[e]; mx = fmaxf(mx, acc[e]); }
    float s = 0.f;
#pragma unroll
    for (int e = 0; e < NEXP; ++e) { acc[e] = __expf(acc[e] - mx); s += acc[e]; }
    float inv = 1.f / s;
#pragma unroll
    for (int e = 0; e < NEXP; ++e) {
      float w = acc[e] * inv;
      rw[(long)t * NEXP + e] = w;
      atomicAdd(&usage[e], w);
    }
  }
}

__global__ void moe_auxloss(const float* __restrict__ usage,
                            float* __restrict__ out_aux) {
  if (threadIdx.x == 0 && blockIdx.x == 0) {
    float s = 0.f;
#pragma unroll
    for (int e = 0; e < NEXP; ++e) {
      float u = usage[e] * (1.f / (float)NTOK);
      float d = u - (1.f / (float)NEXP);
      s += d * d;
    }
    *out_aux = (s / (float)NEXP) * LBW;
  }
}

// ---------------------------------------------------------------------------
// WMMA fragment loaders (16-bit layouts per CDNA5 ISA 7.12.2, wave32).
// ---------------------------------------------------------------------------
__device__ __forceinline__ v16bf moe_load_a(const bf16_t* __restrict__ p) {
  v8bf lo = *(const v8bf*)(p);
  v8bf hi = *(const v8bf*)(p + 16);
  v16bf a;
#pragma unroll
  for (int i = 0; i < 8; ++i) { a[i] = lo[i]; a[i + 8] = hi[i]; }
  return a;
}
__device__ __forceinline__ v16bf moe_lds_b(const char* p) {
  v8bf lo = *(const v8bf*)(p);
  v8bf hi = *(const v8bf*)(p + 32);
  v16bf b;
#pragma unroll
  for (int i = 0; i < 8; ++i) { b[i] = lo[i]; b[i + 8] = hi[i]; }
  return b;
}

// ---------------------------------------------------------------------------
// TDM: async-copy one 128(N) x 64(K) bf16 tile (N-major, 128B rows) from
// global to LDS with +16B/row padding. D# per CDNA5 ISA.
// Issued by wave 0 only; completion via TENSORcnt.
// ---------------------------------------------------------------------------
__device__ __forceinline__ void moe_tdm_load_b(unsigned lds_off,
                                               unsigned long long gaddr,
                                               unsigned k_units) {
  u32x4 g0;
  g0[0] = 1u;                                    // count=1 valid descriptor
  g0[1] = lds_off;                               // LDS byte address
  g0[2] = (unsigned)(gaddr & 0xffffffffu);       // global_addr[31:0]
  g0[3] = (unsigned)((gaddr >> 32) & 0x01ffffffu) | (2u << 30);  // [56:32]|type=2
  i32x8 g1;
  g1[0] = (3 << 16)        // data_size = 8B
        | (1 << 20)        // pad_enable
        | (4 << 22)        // pad_interval: 32 dwords (one 128B row)
        | (3 << 25);       // pad_amount: 4 dwords (16B)
  g1[1] = (16 << 16);      // tensor_dim0 = 16 units
  g1[2] = (128 << 16);     // tensor_dim1 = 128
  g1[3] = (16 << 16);      // tile_dim0 = 16 units
  g1[4] = 128;             // tile_dim1 = 128 rows
  g1[5] = (int)k_units;    // tensor_dim0_stride (row stride, 8B units)
  g1[6] = 0;
  g1[7] = 0;
  i32x4 z4 = {0, 0, 0, 0};
  i32x8 z8 = {0, 0, 0, 0, 0, 0, 0, 0};
  __builtin_amdgcn_tensor_load_to_lds(g0, g1, z4, z4, z8, 0);
}

// ---------------------------------------------------------------------------
// Shared WMMA mainloop: acc[8] = A_tile @ B_tile^T, B via TDM->LDS.
// B fragments explicitly double-buffered: tile nt+1's ds_loads issue before
// tile nt's WMMA so the pre-WMMA wait has slack (dscnt<=2) instead of 0.
// ---------------------------------------------------------------------------
__device__ __forceinline__ void moe_mainloop(
    const bf16_t* __restrict__ ap, const bf16_t* __restrict__ btile,
    int K, int lane, int wave, char* smem, v8f acc[8]) {
  int l16 = lane & 15;
  int bko = (lane & 16) ? 8 : 0;
  unsigned k_units = (unsigned)(K >> 2);
  int nch = K >> 6;
  int fbase = l16 * BROW + bko * 2;   // lane-local fragment byte offset

  if (wave == 0)
    moe_tdm_load_b(0u, (unsigned long long)(uintptr_t)btile, k_units);

  for (int kc = 0; kc < nch; ++kc) {
    if (wave == 0) {
      if (kc + 1 < nch) {
        moe_tdm_load_b((unsigned)(((kc + 1) & 1) * BUFB),
                       (unsigned long long)(uintptr_t)(btile + (long)(kc + 1) * KC),
                       k_units);
        __builtin_amdgcn_s_wait_tensorcnt(1);   // oldest (chunk kc) done
      } else {
        __builtin_amdgcn_s_wait_tensorcnt(0);
      }
    }
    __syncthreads();                            // LDS chunk kc visible to all
    const char* bb = smem + (kc & 1) * BUFB + fbase;
    v16bf bcur = moe_lds_b(bb);                 // fragment (ks=0, nt=0)
#pragma unroll
    for (int ks = 0; ks < 2; ++ks) {
      v16bf a = moe_load_a(ap);
      ap += 32;
      __builtin_prefetch(ap + 64, 0, 1);
#pragma unroll
      for (int nt = 0; nt < 8; ++nt) {
        v16bf bnext = bcur;
        int nn = nt + 1, kn = ks;
        if (nn == 8) { nn = 0; kn = ks + 1; }
        if (kn < 2)                             // issue next frag's ds_loads
          bnext = moe_lds_b(bb + nn * 16 * BROW + kn * 64);
        acc[nt] = __builtin_amdgcn_wmma_f32_16x16x32_bf16(
            false, a, false, bcur, (short)0, acc[nt], false, false);
        bcur = bnext;
      }
    }
    __syncthreads();                            // all reads done before reuse
  }
}

// ---------------------------------------------------------------------------
// FFN1 (gridDim.z = 9): z<8 -> expert z, z==8 -> shared branch.
//   hall[t, z*HID + h] = scale(t,z) * GELU(x @ W1_z + b1_z)
//   scale = rw[t,z] (experts) or 0.1 (shared)  -- router weight folded in.
// ---------------------------------------------------------------------------
__global__ __launch_bounds__(256) void moe_ffn1(
    const bf16_t* __restrict__ A, const bf16_t* __restrict__ w1all,
    const float* __restrict__ b1, const float* __restrict__ bs1,
    const float* __restrict__ rw, bf16_t* __restrict__ hall) {
  extern __shared__ char smem[];
  int lane = threadIdx.x & 31;
  int wave = threadIdx.x >> 5;
  int z = blockIdx.z;
  int m_base = blockIdx.y * 128 + wave * 16;
  int n_base = blockIdx.x * 128;
  int l16 = lane & 15;
  int ako = (lane & 16) ? 8 : 0;

  const bf16_t* ap = A + (long)(m_base + l16) * DIM + ako;
  const bf16_t* btile = w1all + (long)z * HID * DIM + (long)n_base * DIM;

  v8f acc[8] = {};
  moe_mainloop(ap, btile, DIM, lane, wave, smem, acc);

  const float* bias = (z < NEXP) ? (b1 + (long)z * HID) : bs1;
  int mofs = (lane & 16) ? 8 : 0;
  long ocol = (long)z * HID;
  float rsc[8];
#pragma unroll
  for (int r = 0; r < 8; ++r) {
    int row = m_base + r + mofs;
    rsc[r] = (z < NEXP) ? rw[(long)row * NEXP + z] : 0.1f;
  }
#pragma unroll
  for (int nt = 0; nt < 8; ++nt) {
    int col = n_base + nt * 16 + l16;
    float bv = bias[col];
#pragma unroll
    for (int r = 0; r < 8; ++r) {
      int row = m_base + r + mofs;
      float v = acc[nt][r] + bv;
      float g = 0.5f * v * (1.f + erff(v * 0.70710678118f));
      hall[(long)row * KALL + ocol + col] = (bf16_t)(g * rsc[r]);
    }
  }
}

// ---------------------------------------------------------------------------
// FFN2: single stacked GEMM, K = 9216.
//   out[t,d] = hall[t,:] @ w2all[d,:] + sum_e rw[t,e]*b2[e,d] + 0.1*bs2[d]
// Writes out exactly once (no RMW accumulation).
// ---------------------------------------------------------------------------
__global__ __launch_bounds__(256) void moe_ffn2(
    const bf16_t* __restrict__ hall, const bf16_t* __restrict__ w2all,
    const float* __restrict__ b2, const float* __restrict__ bs2,
    const float* __restrict__ rw, float* __restrict__ Out) {
  extern __shared__ char smem[];
  int lane = threadIdx.x & 31;
  int wave = threadIdx.x >> 5;
  int m_base = blockIdx.y * 128 + wave * 16;
  int n_base = blockIdx.x * 128;
  int l16 = lane & 15;
  int ako = (lane & 16) ? 8 : 0;

  const bf16_t* ap = hall + (long)(m_base + l16) * KALL + ako;
  const bf16_t* btile = w2all + (long)n_base * KALL;

  v8f acc[8] = {};
  moe_mainloop(ap, btile, KALL, lane, wave, smem, acc);

  int mofs = (lane & 16) ? 8 : 0;
  float wrow[8][NEXP];            // router weights for this wave's 8 rows
#pragma unroll
  for (int r = 0; r < 8; ++r) {
    int row = m_base + r + mofs;
#pragma unroll
    for (int e = 0; e < NEXP; ++e) wrow[r][e] = rw[(long)row * NEXP + e];
  }
#pragma unroll
  for (int nt = 0; nt < 8; ++nt) {
    int col = n_base + nt * 16 + l16;
    float b2v[NEXP];
#pragma unroll
    for (int e = 0; e < NEXP; ++e) b2v[e] = b2[(long)e * DIM + col];
    float bsh = 0.1f * bs2[col];
#pragma unroll
    for (int r = 0; r < 8; ++r) {
      int row = m_base + r + mofs;
      float bsum = bsh;
#pragma unroll
      for (int e = 0; e < NEXP; ++e) bsum += wrow[r][e] * b2v[e];
      Out[(long)row * DIM + col] = acc[nt][r] + bsum;
    }
  }
}

// ---------------------------------------------------------------------------
// Host launcher
// ---------------------------------------------------------------------------
extern "C" void kernel_launch(void* const* d_in, const int* in_sizes, int n_in,
                              void* d_out, int out_size, void* d_ws, size_t ws_size,
                              hipStream_t stream) {
  (void)in_sizes; (void)n_in; (void)out_size; (void)ws_size;
  const float* x   = (const float*)d_in[0];
  const float* Wr  = (const float*)d_in[1];
  const float* br  = (const float*)d_in[2];
  const float* W1  = (const float*)d_in[3];
  const float* b1  = (const float*)d_in[4];
  const float* W2  = (const float*)d_in[5];
  const float* b2  = (const float*)d_in[6];
  const float* Ws1 = (const float*)d_in[7];
  const float* bs1 = (const float*)d_in[8];
  const float* Ws2 = (const float*)d_in[9];
  const float* bs2 = (const float*)d_in[10];
  float* out = (float*)d_out;

  // workspace carve-up (~250 MB)
  char* p = (char*)d_ws;
  bf16_t* xb    = (bf16_t*)p; p += (size_t)NTOK * DIM * 2;          //  32 MB
  bf16_t* w1all = (bf16_t*)p; p += (size_t)9 * HID * DIM * 2;       //  36 MB [z][h][d]
  bf16_t* w2all = (bf16_t*)p; p += (size_t)DIM * KALL * 2;          //  36 MB [d][e*H+h | 8192+h]
  bf16_t* hall  = (bf16_t*)p; p += (size_t)NTOK * KALL * 2;         // 144 MB
  float*  rw    = (float*)p;  p += (size_t)NTOK * NEXP * 4;
  float*  usage = (float*)p;  p += 64;

  (void)hipMemsetAsync(usage, 0, NEXP * sizeof(float), stream);

  // Stage 1: precision staging (fp32 -> bf16, weights transposed to N-major)
  {
    int n = NTOK * DIM;
    moe_cvt_bf16<<<n / 8 / 256, 256, 0, stream>>>(x, xb, n);
  }
  dim3 tb(32, 8);
  // W1 (E,D,H) -> w1all[z][h][d]: per-batch block, ostride=DIM, obatch=HID*DIM
  moe_xpose_cvt<<<dim3(HID / 32, DIM / 32, NEXP), tb, 0, stream>>>(
      W1, w1all, DIM, HID, DIM, (long)HID * DIM);
  // Ws1 (D,HID) -> w1all block z=8
  moe_xpose_cvt<<<dim3(HID / 32, DIM / 32, 1), tb, 0, stream>>>(
      Ws1, w1all + (size_t)NEXP * HID * DIM, DIM, HID, DIM, 0);
  // W2 (E,H,D) -> w2all[d][e*HID+h]: column-interleaved, ostride=KALL, obatch=HID
  moe_xpose_cvt<<<dim3(DIM / 32, HID / 32, NEXP), tb, 0, stream>>>(
      W2, w2all, HID, DIM, KALL, (long)HID);
  // Ws2 (HID,D) -> w2all columns [8192..9216)
  moe_xpose_cvt<<<dim3(DIM / 32, HID / 32, 1), tb, 0, stream>>>(
      Ws2, w2all + (size_t)NEXP * HID, HID, DIM, KALL, 0);

  // Stage 2: router + aux loss
  moe_router<<<NTOK / 8, 256, 0, stream>>>(x, Wr, br, rw, usage);
  moe_auxloss<<<1, 32, 0, stream>>>(usage, out + (size_t)NTOK * DIM);

  // Stage 3: FFN1 — all 8 experts + shared branch in one launch (grid.z=9),
  // router weight / 0.1 folded into hall.
  moe_ffn1<<<dim3(HID / 128, NTOK / 128, NEXP + 1), 256, SMEMB, stream>>>(
      xb, w1all, b1, bs1, rw, hall);

  // Stage 4: FFN2 — single stacked K=9216 GEMM, out written once.
  moe_ffn2<<<dim3(DIM / 128, NTOK / 128), 256, SMEMB, stream>>>(
      hall, w2all, b2, bs2, rw, out);
}
